// BaseModel_79250736546631
// MI455X (gfx1250) — compile-verified, matrix operand order checked by reading, not run on hardware
//
#include <hip/hip_runtime.h>
#include <hip/hip_fp16.h>

typedef __attribute__((ext_vector_type(16))) _Float16 v16h;
typedef __attribute__((ext_vector_type(8)))  float    v8f;

#define EMB    64
#define HIST   200
#define HISTP1 201
#define DIN    128   // 2*EMB
#define H1     120
#define H2     60

// workspace layout (bytes):
//   [0, 4MB)              h   : B x 128 f16 activations
//   [4MB, 4MB+32KB)       W1p : 128 cols x 128 k, f16, zero-padded
//   [4MB+32KB, +16KB)     W2p : 64 cols x 128 k, f16, zero-padded
#define W1P_ELEMS (128 * 128)
#define W2P_ELEMS (64 * 128)

// ---------------------------------------------------------------------------
// Kernel 0: pack W1/W2 (f32 row-major KxN) into f16 col-major [N][K] with
// zero padding, so WMMA B-fragments are contiguous unconditional loads.
// ---------------------------------------------------------------------------
__global__ __launch_bounds__(256) void din_pack(
    const float* __restrict__ W1, const float* __restrict__ W2,
    _Float16* __restrict__ W1p, _Float16* __restrict__ W2p)
{
    const int i = blockIdx.x * 256 + threadIdx.x;
    if (i < W1P_ELEMS) {
        const int c = i >> 7, k = i & 127;
        W1p[i] = (c < H1) ? (_Float16)W1[k * H1 + c] : (_Float16)0.f;
    } else if (i < W1P_ELEMS + W2P_ELEMS) {
        const int j = i - W1P_ELEMS;
        const int c = j >> 7, k = j & 127;
        W2p[j] = (c < H2 && k < H1) ? (_Float16)W2[k * H2 + c] : (_Float16)0.f;
    }
}

// ---------------------------------------------------------------------------
// Kernel 1: embedding gather + weighted sum  ->  h[B][128] in f16
//   user_vec = (cnt_nonzero/200) * sum_{t: x>0} emb[x[t]]   (dims 0..63)
//   ad_emb   = emb[x[200]]                                   (dims 64..127)
// One block per batch row; 256 threads = 4 sub-streams x 64 dims.
// Branch-free inner loop: unconditional coalesced 256B row reads, masked FMA.
// ---------------------------------------------------------------------------
__global__ __launch_bounds__(256) void din_gather(
    const int* __restrict__ x,
    const float* __restrict__ emb,
    _Float16* __restrict__ h)
{
    const int row = blockIdx.x;
    const int tid = threadIdx.x;
    const int* xrow = x + (long)row * HISTP1;

    __shared__ int   s_cnt;
    __shared__ float s_red[256];

    if (tid == 0) s_cnt = 0;
    __syncthreads();
    if (tid < HIST) {
        if (xrow[tid] > 0) atomicAdd(&s_cnt, 1);
    }
    __syncthreads();
    const float avg = (float)s_cnt * (1.0f / (float)HIST);

    const int d   = tid & 63;   // embedding dim
    const int sub = tid >> 6;   // history sub-stream 0..3
    float acc = 0.f;
    for (int t = sub; t < HIST; t += 4) {
        const int   idx = xrow[t];                      // always >= 0
        const float m   = (idx > 0) ? 1.f : 0.f;
        acc = fmaf(m, emb[(long)idx * EMB + d], acc);   // unconditional load
    }
    s_red[tid] = acc;
    __syncthreads();

    if (sub == 0) {
        const float s = s_red[d] + s_red[64 + d] + s_red[128 + d] + s_red[192 + d];
        h[(long)row * DIN + d] = (_Float16)(s * avg);
    } else if (sub == 1) {
        const int ad = xrow[HIST];
        h[(long)row * DIN + EMB + d] = (_Float16)emb[(long)ad * EMB + d];
    }
}

// ---------------------------------------------------------------------------
// Kernel 2: MLP 128 -> 120 (relu) -> 60 (relu) -> 1 (sigmoid) via WMMA f16.
// One block = 16 batch rows, 128 threads = 4 waves.
// Fragment layout (ISA 7.12.2, 16-bit 16x32): per lane, two contiguous
// 8-f16 runs at k = hlf*8 and k = 16 + hlf*8  ->  two b128 loads each.
// ---------------------------------------------------------------------------
union AF { v16h v; uint4 q[2]; };

__global__ __launch_bounds__(128) void din_mlp(
    const _Float16* __restrict__ h,
    const _Float16* __restrict__ W1p, const float* __restrict__ b1,
    const _Float16* __restrict__ W2p, const float* __restrict__ b2,
    const float* __restrict__ W3,     const float* __restrict__ b3,
    float* __restrict__ out)
{
    const int row0 = blockIdx.x * 16;
    const int tid  = threadIdx.x;
    const int wave = tid >> 5;     // 0..3
    const int lane = tid & 31;
    const int hlf  = lane >> 4;    // 0/1
    const int l16  = lane & 15;

    __shared__ _Float16 s_h1[16][DIN];  // layer-1 out, zero-padded to 128 cols
    __shared__ _Float16 s_h2[16][64];   // layer-2 out, zero-padded to 64 cols

    // ---------------- layer 1: h(16x128) @ W1(128x120), N-tiles {w, w+4} ----
    const _Float16* arow = h + (long)(row0 + l16) * DIN;     // A row (M = l16)
    #pragma unroll
    for (int it = 0; it < 2; ++it) {
        const int nt  = wave + 4 * it;                       // uniform per wave
        const int col = nt * 16 + l16;
        const _Float16* brow = W1p + (size_t)col * 128;      // packed B column
        v8f c = {};
        #pragma unroll
        for (int kk = 0; kk < 4; ++kk) {
            AF a, b;
            a.q[0] = *(const uint4*)(arow + kk * 32 + hlf * 8);
            a.q[1] = *(const uint4*)(arow + kk * 32 + 16 + hlf * 8);
            b.q[0] = *(const uint4*)(brow + kk * 32 + hlf * 8);
            b.q[1] = *(const uint4*)(brow + kk * 32 + 16 + hlf * 8);
            c = __builtin_amdgcn_wmma_f32_16x16x32_f16(false, a.v, false, b.v,
                                                       (short)0, c, false, false);
        }
        // fused bias + relu -> LDS f16 (value is 0 for padded cols >= 120)
        const float bias = (col < H1) ? b1[col] : 0.f;
        #pragma unroll
        for (int r = 0; r < 8; ++r) {
            const int m = r + 8 * hlf;          // C/D layout: M = r + 8*(lane/16)
            float v = c[r] + bias;
            v = (col < H1 && v > 0.f) ? v : 0.f;
            s_h1[m][col] = (_Float16)v;
        }
    }
    __syncthreads();

    // ---------------- layer 2: h1(16x128pad) @ W2p(128pad x 64pad) ----------
    {
        const int col = wave * 16 + l16;                     // N-tile = wave
        const _Float16* brow = W2p + (size_t)col * 128;
        v8f c = {};
        #pragma unroll
        for (int kk = 0; kk < 4; ++kk) {
            AF a, b;
            a.q[0] = *(const uint4*)&s_h1[l16][kk * 32 + hlf * 8];
            a.q[1] = *(const uint4*)&s_h1[l16][kk * 32 + 16 + hlf * 8];
            b.q[0] = *(const uint4*)(brow + kk * 32 + hlf * 8);
            b.q[1] = *(const uint4*)(brow + kk * 32 + 16 + hlf * 8);
            c = __builtin_amdgcn_wmma_f32_16x16x32_f16(false, a.v, false, b.v,
                                                       (short)0, c, false, false);
        }
        const float bias = (col < H2) ? b2[col] : 0.f;
        #pragma unroll
        for (int r = 0; r < 8; ++r) {
            const int m = r + 8 * hlf;
            float v = c[r] + bias;
            v = (col < H2 && v > 0.f) ? v : 0.f;
            s_h2[m][col] = (_Float16)v;
        }
    }
    __syncthreads();

    // ---------------- layer 3: 60-dot + bias + sigmoid ----------------
    if (tid < 16) {
        float dot = b3[0];
        #pragma unroll 4
        for (int k = 0; k < H2; ++k)
            dot += (float)s_h2[tid][k] * W3[k];
        out[row0 + tid] = 1.f / (1.f + __expf(-dot));
    }
}

// ---------------------------------------------------------------------------
extern "C" void kernel_launch(void* const* d_in, const int* in_sizes, int n_in,
                              void* d_out, int out_size, void* d_ws, size_t ws_size,
                              hipStream_t stream)
{
    const int*   x   = (const int*)d_in[0];
    const float* emb = (const float*)d_in[1];
    const float* W1  = (const float*)d_in[2];
    const float* b1  = (const float*)d_in[3];
    const float* W2  = (const float*)d_in[4];
    const float* b2  = (const float*)d_in[5];
    const float* W3  = (const float*)d_in[6];
    const float* b3  = (const float*)d_in[7];
    float* out = (float*)d_out;

    const int B = in_sizes[0] / HISTP1;                 // 16384
    char* ws = (char*)d_ws;
    _Float16* h   = (_Float16*)ws;                      // B*128 f16 = 4 MB
    _Float16* W1p = (_Float16*)(ws + (size_t)B * DIN * sizeof(_Float16));
    _Float16* W2p = W1p + W1P_ELEMS;

    din_pack<<<(W1P_ELEMS + W2P_ELEMS + 255) / 256, 256, 0, stream>>>(W1, W2, W1p, W2p);
    din_gather<<<B, 256, 0, stream>>>(x, emb, h);
    din_mlp<<<B / 16, 128, 0, stream>>>(h, W1p, b1, W2p, b2, W3, b3, out);
}
